// TreeRNN_9620726743762
// MI455X (gfx1250) — matrix operand order repaired and checked by reading, not compiled, not run on hardware
//
#include <hip/hip_runtime.h>

// TreeRNN on gfx1250 (MI455X): level-synchronous scan with WMMA f16 matvecs.
//
// N=8192 nodes, DIM=100 (out padded ->112=7*16, K padded ->128=4*32), L=40
// labels, LMAX=64 levels. Latency-bound: per-level kernels on the stream form
// the dependency chain; per-node matvec W[etype]*h is a 16x16x32 WMMA with h
// broadcast across all 16 B-columns. All activation vectors are staged as
// zero-padded f16 rows of stride 128 so every WMMA operand is built from
// aligned b128 loads (no bounds-predicated gathers).

#define N_NODES 8192
#define DIM     100
#define NLAB    40
#define LMAX    64
#define DIMP    112   // padded output dim (7 * 16)
#define KP      128   // padded inner dim  (4 * 32) == row stride everywhere

typedef __attribute__((ext_vector_type(16))) _Float16 v16h;
typedef __attribute__((ext_vector_type(8)))  _Float16 v8h;
typedef __attribute__((ext_vector_type(8)))  float    v8f;

// ---------------------------------------------------------------------------
// WMMA operand builders (wave32; A/B use this code's contiguous k order, not the ISA table order)
// A (16xK f16): lane 0-15 row M=lane, elems 0..15 = K 0..15
//               lane 16-31 row M=lane-16, elems 0..15 = K 16..31
// B (Kx16 f16): lane 0-15 col N=lane, elems = K 0..15; lanes 16-31 K 16..31
// D (16x16 f32): vgpr r, lane l -> M = r + (l<16?0:8), N = l&15
// ---------------------------------------------------------------------------

__device__ __forceinline__ v16h load_A_f16(const _Float16* __restrict__ row,
                                           int kstep, int lane) {
  const int base = (lane < 16) ? 0 : 16;
  const v8h lo = *(const v8h*)(row + kstep * 32 + base);        // K base..base+7
  const v8h hi = *(const v8h*)(row + kstep * 32 + base + 8);    // K base+8..base+15
  v16h a;
#pragma unroll
  for (int e = 0; e < 8; ++e) { a[e] = lo[e]; a[e + 8] = hi[e]; }
  return a;
}

__device__ __forceinline__ v16h load_B_f16(const _Float16* __restrict__ colrow,
                                           int kstep, int lane) {
  // 32B-aligned contiguous 16 halves: K = kstep*32 + (lane<16 ? 0..15 : 16..31)
  return *(const v16h*)(colrow + kstep * 32 + ((lane < 16) ? 0 : 16));
}

// ---------------------------------------------------------------------------
// Weight conversion: 82 matrices (Wv, W[0..39], Dm[0..39], Dv) -> f16 padded
// row-major [o(112)][i(128)], zeros in pad region.
// ---------------------------------------------------------------------------
__global__ void prep_weights(const float* __restrict__ Wv, const float* __restrict__ W,
                             const float* __restrict__ Dm, const float* __restrict__ Dv,
                             _Float16* __restrict__ wf16) {
  const int per = DIMP * KP;
  int idx = blockIdx.x * blockDim.x + threadIdx.x;
  if (idx >= 82 * per) return;
  const int m = idx / per;
  const int r = idx % per;
  const int o = r / KP, i = r % KP;
  float v = 0.0f;
  if (o < DIM && i < DIM) {
    if (m == 0)            v = Wv[o * DIM + i];
    else if (m <= NLAB)    v = W[(m - 1) * DIM * DIM + o * DIM + i];
    else if (m <= 2*NLAB)  v = Dm[(m - 1 - NLAB) * DIM * DIM + o * DIM + i];
    else                   v = Dv[o * DIM + i];
  }
  wf16[idx] = (_Float16)v;
}

// ---------------------------------------------------------------------------
// Pack x (N x 100 f32) -> Xf16 (N x 128 f16, zero pad)
// ---------------------------------------------------------------------------
__global__ void pack_x(const float* __restrict__ x, _Float16* __restrict__ xf16) {
  int idx = blockIdx.x * blockDim.x + threadIdx.x;
  if (idx >= N_NODES * KP) return;
  const int n = idx >> 7, i = idx & (KP - 1);
  xf16[idx] = (i < DIM) ? (_Float16)x[n * DIM + i] : (_Float16)0.0f;
}

// ---------------------------------------------------------------------------
// xWv = x @ Wv.T + bWv   -> padded f32 (N x 128), one wave per 16x16 tile
// ---------------------------------------------------------------------------
__global__ void xwv_kernel(const _Float16* __restrict__ xf16,
                           const _Float16* __restrict__ wv16,
                           const float* __restrict__ bWv, float* __restrict__ xwv) {
  const int wg = blockIdx.x * (blockDim.x >> 5) + (threadIdx.x >> 5);
  const int lane = threadIdx.x & 31;
  const int OT = DIMP / 16;                       // 7 output tiles
  if (wg >= (N_NODES / 16) * OT) return;
  const int nt = wg / OT, ot = wg % OT;
  const _Float16* xrow = xf16 + (size_t)(nt * 16 + (lane & 15)) * KP;
  const _Float16* wrow = wv16 + (size_t)(ot * 16 + (lane & 15)) * KP;
  v8f acc = {};
#pragma unroll
  for (int k = 0; k < 4; ++k) {
    v16h a = load_A_f16(xrow, k, lane);
    v16h b = load_B_f16(wrow, k, lane);
    acc = __builtin_amdgcn_wmma_f32_16x16x32_f16(false, a, false, b,
                                                 (short)0, acc, false, false);
  }
  const int col = ot * 16 + (lane & 15);
  const float bias = (col < DIM) ? bWv[col] : 0.0f;
  const int mof = (lane < 16) ? 0 : 8;
#pragma unroll
  for (int r = 0; r < 8; ++r) {
    xwv[(size_t)(nt * 16 + mof + r) * KP + col] = acc[r] + bias;
  }
}

// ---------------------------------------------------------------------------
// Synthesis, one level: for node at depth d:
//   h = tanh(xWv[n] + S[n]); Hf16[n] = h;
//   if d>0: S[parent[n]] += W[etype[n]] @ h + bW[etype[n]]   (atomic f32)
// One wave per node; h staged in LDS (f16) as broadcast-B for WMMA.
// ---------------------------------------------------------------------------
__global__ void synth_level(int d, const float* __restrict__ xwv, float* __restrict__ S,
                            _Float16* __restrict__ Hf16, const _Float16* __restrict__ w16,
                            const float* __restrict__ bW, const int* __restrict__ parent,
                            const int* __restrict__ etype, const int* __restrict__ depth) {
  __shared__ __attribute__((aligned(32))) _Float16 hbuf[8 * KP];
  const int wave = threadIdx.x >> 5, lane = threadIdx.x & 31;
  const int node = blockIdx.x * 8 + wave;
  if (node >= N_NODES) return;
  if (depth[node] != d) return;                   // wave-uniform: EXEC stays all-1s

  _Float16* hl = hbuf + wave * KP;
#pragma unroll
  for (int i = lane; i < KP; i += 32) {
    float v = 0.0f;
    if (i < DIMP)
      v = tanhf(xwv[(size_t)node * KP + i] + S[(size_t)node * KP + i]);
    const _Float16 hv = (_Float16)v;
    hl[i] = hv;
    Hf16[(size_t)node * KP + i] = hv;
  }
  if (d == 0) return;

  const int l = etype[node];
  const int p = parent[node];
  const _Float16* wl = w16 + (size_t)l * DIMP * KP;
#pragma unroll 1
  for (int t = 0; t < 7; ++t) {
    const _Float16* arow = wl + (size_t)(t * 16 + (lane & 15)) * KP;
    v8f acc = {};
#pragma unroll
    for (int k = 0; k < 4; ++k) {
      v16h a = load_A_f16(arow, k, lane);
      v16h b = load_B_f16(hl, k, lane);                     // h broadcast to 16 cols
      acc = __builtin_amdgcn_wmma_f32_16x16x32_f16(false, a, false, b,
                                                   (short)0, acc, false, false);
    }
    if (lane == 0 || lane == 16) {                          // cols identical; take N=0
      const int mof = (lane == 0) ? 0 : 8;
#pragma unroll
      for (int r = 0; r < 8; ++r) {
        const int o = t * 16 + mof + r;
        if (o < DIM) atomicAdd(&S[(size_t)p * KP + o], acc[r] + bW[l * DIM + o]);
      }
    }
  }
}

// ---------------------------------------------------------------------------
// Reconstruction base: Uf16 = (depth==0) ? Hf16 : 0   (full N x 128)
// ---------------------------------------------------------------------------
__global__ void recon_base(const _Float16* __restrict__ Hf16, _Float16* __restrict__ Uf16,
                           const int* __restrict__ depth) {
  int idx = blockIdx.x * blockDim.x + threadIdx.x;
  if (idx >= N_NODES * KP) return;
  Uf16[idx] = (depth[idx >> 7] == 0) ? Hf16[idx] : (_Float16)0.0f;
}

// ---------------------------------------------------------------------------
// Reconstruction, one level: U[n] = tanh(Dm[etype[n]] @ U[parent[n]] + bDm)
// Parent row is read straight from global as the broadcast-B operand.
// ---------------------------------------------------------------------------
__global__ void recon_level(int d, _Float16* __restrict__ Uf16,
                            const _Float16* __restrict__ dm16,
                            const float* __restrict__ bDm, const int* __restrict__ parent,
                            const int* __restrict__ etype, const int* __restrict__ depth) {
  const int wave = threadIdx.x >> 5, lane = threadIdx.x & 31;
  const int node = blockIdx.x * 8 + wave;
  if (node >= N_NODES) return;
  if (depth[node] != d) return;

  const int p = parent[node];
  const int l = etype[node];
  const _Float16* ul = Uf16 + (size_t)p * KP;     // padded row, pad cols are 0
  const _Float16* ml = dm16 + (size_t)l * DIMP * KP;
#pragma unroll 1
  for (int t = 0; t < 7; ++t) {
    const _Float16* arow = ml + (size_t)(t * 16 + (lane & 15)) * KP;
    v8f acc = {};
#pragma unroll
    for (int k = 0; k < 4; ++k) {
      v16h a = load_A_f16(arow, k, lane);
      v16h b = load_B_f16(ul, k, lane);
      acc = __builtin_amdgcn_wmma_f32_16x16x32_f16(false, a, false, b,
                                                   (short)0, acc, false, false);
    }
    if (lane == 0 || lane == 16) {
      const int mof = (lane == 0) ? 0 : 8;
#pragma unroll
      for (int r = 0; r < 8; ++r) {
        const int o = t * 16 + mof + r;
        const float bias = (o < DIM) ? bDm[l * DIM + o] : 0.0f;
        Uf16[(size_t)node * KP + o] = (_Float16)tanhf(acc[r] + bias); // pad o -> 0
      }
    }
  }
}

// ---------------------------------------------------------------------------
// R = is_child ? tanh(U @ Dv.T + bDv) : 0   -> compact (N x 100) f32 output
// ---------------------------------------------------------------------------
__global__ void final_kernel(const _Float16* __restrict__ Uf16,
                             const _Float16* __restrict__ dv16,
                             const float* __restrict__ bDv, const int* __restrict__ depth,
                             float* __restrict__ out) {
  const int wg = blockIdx.x * (blockDim.x >> 5) + (threadIdx.x >> 5);
  const int lane = threadIdx.x & 31;
  const int OT = DIMP / 16;
  if (wg >= (N_NODES / 16) * OT) return;
  const int nt = wg / OT, ot = wg % OT;
  const _Float16* urow = Uf16 + (size_t)(nt * 16 + (lane & 15)) * KP;
  const _Float16* wrow = dv16 + (size_t)(ot * 16 + (lane & 15)) * KP;
  v8f acc = {};
#pragma unroll
  for (int k = 0; k < 4; ++k) {
    v16h a = load_A_f16(urow, k, lane);
    v16h b = load_B_f16(wrow, k, lane);
    acc = __builtin_amdgcn_wmma_f32_16x16x32_f16(false, a, false, b,
                                                 (short)0, acc, false, false);
  }
  const int col = ot * 16 + (lane & 15);
  if (col >= DIM) return;
  const float bias = bDv[col];
  const int mof = (lane < 16) ? 0 : 8;
#pragma unroll
  for (int r = 0; r < 8; ++r) {
    const int m = nt * 16 + mof + r;
    out[m * DIM + col] = (depth[m] > 0) ? tanhf(acc[r] + bias) : 0.0f;
  }
}

// ---------------------------------------------------------------------------
// Launch: graph-capturable (memsetAsync + per-level kernels on `stream`).
// ws layout (bytes):
//   S    : N*128 f32   (4 MB, memset 0)
//   XWV  : N*128 f32   (4 MB)
//   Xf16 : N*128 f16   (2 MB)
//   Hf16 : N*128 f16   (2 MB)
//   Uf16 : N*128 f16   (2 MB)
//   wf16 : 82*112*128 f16 (~2.3 MB)   order: Wv, W[0..39], Dm[0..39], Dv
// ---------------------------------------------------------------------------
extern "C" void kernel_launch(void* const* d_in, const int* in_sizes, int n_in,
                              void* d_out, int out_size, void* d_ws, size_t ws_size,
                              hipStream_t stream) {
  (void)in_sizes; (void)n_in; (void)out_size; (void)ws_size;
  const float* x    = (const float*)d_in[0];
  const float* Wv   = (const float*)d_in[1];
  const float* bWv  = (const float*)d_in[2];
  const float* W    = (const float*)d_in[3];
  const float* bW   = (const float*)d_in[4];
  const float* Dv   = (const float*)d_in[5];
  const float* bDv  = (const float*)d_in[6];
  const float* Dm   = (const float*)d_in[7];
  const float* bDm  = (const float*)d_in[8];
  const int* parent = (const int*)d_in[9];
  const int* etype  = (const int*)d_in[10];
  const int* depth  = (const int*)d_in[11];
  float* out = (float*)d_out;

  const size_t f32Bytes = (size_t)N_NODES * KP * sizeof(float);
  const size_t f16Bytes = (size_t)N_NODES * KP * sizeof(_Float16);
  char* ws = (char*)d_ws;
  float* S        = (float*)ws;     ws += f32Bytes;
  float* XWV      = (float*)ws;     ws += f32Bytes;
  _Float16* Xf16  = (_Float16*)ws;  ws += f16Bytes;
  _Float16* Hf16  = (_Float16*)ws;  ws += f16Bytes;
  _Float16* Uf16  = (_Float16*)ws;  ws += f16Bytes;
  _Float16* wf16  = (_Float16*)ws;  // 82 * 112 * 128 halves
  _Float16* wv16 = wf16;
  _Float16* w16  = wf16 + (size_t)1 * DIMP * KP;
  _Float16* dm16 = wf16 + (size_t)(1 + NLAB) * DIMP * KP;
  _Float16* dv16 = wf16 + (size_t)(1 + 2 * NLAB) * DIMP * KP;

  hipMemsetAsync(S, 0, f32Bytes, stream);

  prep_weights<<<(82 * DIMP * KP + 255) / 256, 256, 0, stream>>>(Wv, W, Dm, Dv, wf16);
  pack_x<<<(N_NODES * KP + 255) / 256, 256, 0, stream>>>(x, Xf16);

  const int gemmBlocks = ((N_NODES / 16) * (DIMP / 16) + 7) / 8;  // 8 waves/block
  xwv_kernel<<<gemmBlocks, 256, 0, stream>>>(Xf16, wv16, bWv, XWV);

  for (int d = LMAX - 1; d >= 0; --d)
    synth_level<<<N_NODES / 8, 256, 0, stream>>>(d, XWV, S, Hf16, w16, bW,
                                                 parent, etype, depth);

  recon_base<<<(N_NODES * KP + 255) / 256, 256, 0, stream>>>(Hf16, Uf16, depth);

  for (int d = 1; d < LMAX; ++d)
    recon_level<<<N_NODES / 8, 256, 0, stream>>>(d, Uf16, dm16, bDm,
                                                 parent, etype, depth);

  final_kernel<<<gemmBlocks, 256, 0, stream>>>(Uf16, dv16, bDv, depth, out);
}
